// GAT_33131377721924
// MI455X (gfx1250) — matrix operand order repaired and checked by reading, compiled, not run on hardware
//
#include <hip/hip_runtime.h>
#include <hip/hip_bf16.h>

typedef __attribute__((ext_vector_type(16))) _Float16 v16h;
typedef __attribute__((ext_vector_type(8)))  float    v8f;

#define NEG_SLOPE 0.2f
#define EPS 1e-9f

// ---------------------------------------------------------------------------
// Generic fill
// ---------------------------------------------------------------------------
__global__ void fill_f32(float* __restrict__ p, float v, int n) {
    int i = blockIdx.x * blockDim.x + threadIdx.x;
    if (i < n) p[i] = v;
}

// ---------------------------------------------------------------------------
// WMMA GEMM: C[nrows x NCOL] = A[nrows x KDIM] * B[KDIM x NCOL]
// f32 inputs converted to f16 fragments, f32 accumulation via
// v_wmma_f32_16x16x32_f16. One wave per 16x16 output tile; blockDim =
// 32 * (NCOL/16) waves cover one row of tiles; grid.x covers M tiles.
// Fragment layouts per CDNA5 ISA 7.12.2 (wave32).
// ---------------------------------------------------------------------------
template <int KDIM, int NCOL>
__global__ __launch_bounds__(32 * (NCOL / 16))
void gat_wmma_gemm(const float* __restrict__ A, const float* __restrict__ B,
                   float* __restrict__ C, int nrows) {
    const int lane  = threadIdx.x & 31;
    const int wave  = threadIdx.x >> 5;       // N-tile index
    const int mtile = blockIdx.x;
    const int halfw = lane >> 4;              // 0: lanes 0-15, 1: lanes 16-31
    const int l16   = lane & 15;

    const int arow  = mtile * 16 + l16;       // A row this lane supplies
    const int bcol  = wave * 16 + l16;        // B/D column this lane supplies
    const int arowc = arow < nrows ? arow : (nrows - 1);   // clamp for tail safety

    const float* __restrict__ Ap = A + (size_t)arowc * KDIM;

    v8f acc = {};
    #pragma unroll
    for (int k0 = 0; k0 < KDIM; k0 += 32) {
        // ---- A fragment: 16-bit A 16x32.  lanes 0-15: K = k0+0..7, k0+16..23
        //                                   lanes 16-31: K = k0+8..15, k0+24..31
        const int kA = k0 + halfw * 8;
        float4 a0 = *reinterpret_cast<const float4*>(Ap + kA);
        float4 a1 = *reinterpret_cast<const float4*>(Ap + kA + 4);
        float4 a2 = *reinterpret_cast<const float4*>(Ap + kA + 16);
        float4 a3 = *reinterpret_cast<const float4*>(Ap + kA + 20);
        float av[16] = {a0.x, a0.y, a0.z, a0.w, a1.x, a1.y, a1.z, a1.w,
                        a2.x, a2.y, a2.z, a2.w, a3.x, a3.y, a3.z, a3.w};
        v16h a;
        #pragma unroll
        for (int j = 0; j < 16; ++j) a[j] = (_Float16)av[j];

        // ---- B fragment: 16-bit B 32x16.  lanes 0-15: K = k0+0..7, k0+16..23
        //                                   lanes 16-31: K = k0+8..15, k0+24..31
        const int kB = k0 + halfw * 8;
        v16h b;
        #pragma unroll
        for (int j = 0; j < 16; ++j)
            b[j] = (_Float16)B[(size_t)(kB + j + (j & 8)) * NCOL + bcol];

        acc = __builtin_amdgcn_wmma_f32_16x16x32_f16(
            /*neg_a=*/false, a, /*neg_b=*/false, b,
            /*c_mod=*/(short)0, acc, /*reuse_a=*/false, /*reuse_b=*/false);
    }

    // ---- D layout: VGPR r -> M = r (lanes 0-15) / 8+r (lanes 16-31), N = l16
    const int mbase = mtile * 16 + halfw * 8;
    #pragma unroll
    for (int r = 0; r < 8; ++r) {
        int row = mbase + r;
        if (row < nrows) C[(size_t)row * NCOL + bcol] = acc[r];
    }
}

// ---------------------------------------------------------------------------
// Per-node attention projections: el[n,h] = dot(feat[n,h,:], a_l[h,:])
// One block per node, one wave per head, wave-level shuffle reduction.
// ---------------------------------------------------------------------------
__global__ void node_attn_proj(const float* __restrict__ feat,
                               const float* __restrict__ al,
                               const float* __restrict__ ar,
                               float* __restrict__ el, float* __restrict__ er,
                               int H) {
    const int n = blockIdx.x;
    const int t = threadIdx.x;            // 0 .. H*32-1
    const int h = t >> 5;
    float v  = feat[(size_t)n * (H * 32) + t];
    float pl = v * al[t];
    float pr = v * ar[t];
    #pragma unroll
    for (int off = 16; off > 0; off >>= 1) {
        pl += __shfl_down(pl, off, 32);
        pr += __shfl_down(pr, off, 32);
    }
    if ((t & 31) == 0) {
        el[(size_t)n * H + h] = pl;
        er[(size_t)n * H + h] = pr;
    }
}

// ---------------------------------------------------------------------------
// Float atomic max via CAS (portable, handles negatives / -inf init)
// ---------------------------------------------------------------------------
__device__ __forceinline__ void atomicMaxF(float* addr, float val) {
    unsigned int* ua = (unsigned int*)addr;
    unsigned int old = *ua;
    while (__uint_as_float(old) < val) {
        unsigned int assumed = old;
        old = atomicCAS(ua, assumed, __float_as_uint(val));
        if (old == assumed) break;
    }
}

__device__ __forceinline__ float leaky(float x) {
    return x > 0.f ? x : NEG_SLOPE * x;
}

// Pass 1: segment max of edge logits over dst
__global__ void edge_logit_max(const float* __restrict__ el,
                               const float* __restrict__ er,
                               const int* __restrict__ src,
                               const int* __restrict__ dst,
                               float* __restrict__ emax, int E, int H) {
    int e = blockIdx.x * blockDim.x + threadIdx.x;
    if (e >= E) return;
    int s = src[e], d = dst[e];
    for (int h = 0; h < H; ++h) {
        float x = leaky(el[(size_t)s * H + h] + er[(size_t)d * H + h]);
        atomicMaxF(&emax[(size_t)d * H + h], x);
    }
}

// Pass 2: ee = exp(logit - emax[dst]); denom[dst] += ee
__global__ void edge_exp_sum(const float* __restrict__ el,
                             const float* __restrict__ er,
                             const int* __restrict__ src,
                             const int* __restrict__ dst,
                             const float* __restrict__ emax,
                             float* __restrict__ ee,
                             float* __restrict__ denom, int E, int H) {
    int e = blockIdx.x * blockDim.x + threadIdx.x;
    if (e >= E) return;
    int s = src[e], d = dst[e];
    for (int h = 0; h < H; ++h) {
        float x  = leaky(el[(size_t)s * H + h] + er[(size_t)d * H + h]);
        float ex = __expf(x - emax[(size_t)d * H + h]);
        ee[(size_t)e * H + h] = ex;
        atomicAdd(&denom[(size_t)d * H + h], ex);
    }
}

// Pass 3: out[dst] += (ee/denom) * feat[src].  One block per edge, H*32 threads.
__global__ void edge_scatter(const float* __restrict__ feat,
                             const float* __restrict__ ee,
                             const float* __restrict__ denom,
                             const int* __restrict__ src,
                             const int* __restrict__ dst,
                             float* __restrict__ out, int H) {
    const int e = blockIdx.x;
    const int t = threadIdx.x;            // 0 .. H*32-1
    const int h = t >> 5;
    const int s = src[e], d = dst[e];
    float alpha = ee[(size_t)e * H + h] / (denom[(size_t)d * H + h] + EPS);
    float v     = feat[(size_t)s * (H * 32) + t];
    atomicAdd(&out[(size_t)d * (H * 32) + t], alpha * v);
}

// ELU(x + bias) in place (F must be power of two)
__global__ void elu_bias(float* __restrict__ x, const float* __restrict__ b,
                         int total, int F) {
    int i = blockIdx.x * blockDim.x + threadIdx.x;
    if (i >= total) return;
    float v = x[i] + b[i & (F - 1)];
    x[i] = v > 0.f ? v : expm1f(v);
}

// out = x + bias (final head-mean over 1 head is identity)
__global__ void add_bias_out(const float* __restrict__ x,
                             const float* __restrict__ b,
                             float* __restrict__ out, int total, int F) {
    int i = blockIdx.x * blockDim.x + threadIdx.x;
    if (i >= total) return;
    out[i] = x[i] + b[i & (F - 1)];
}

// ---------------------------------------------------------------------------
// Host launch
// ---------------------------------------------------------------------------
extern "C" void kernel_launch(void* const* d_in, const int* in_sizes, int n_in,
                              void* d_out, int out_size, void* d_ws, size_t ws_size,
                              hipStream_t stream) {
    const float* h    = (const float*)d_in[0];   // [N,256]
    const float* W1   = (const float*)d_in[1];   // [256,128]
    const float* a_l1 = (const float*)d_in[2];   // [4,32]
    const float* a_r1 = (const float*)d_in[3];
    const float* b1   = (const float*)d_in[4];   // [4,32]
    const float* W2   = (const float*)d_in[5];   // [128,32]
    const float* a_l2 = (const float*)d_in[6];   // [1,32]
    const float* a_r2 = (const float*)d_in[7];
    const float* b2   = (const float*)d_in[8];   // [1,32]
    const int*   src  = (const int*)d_in[9];     // [E]
    const int*   dst  = (const int*)d_in[10];    // [E]

    const int N = in_sizes[0] / 256;             // 50000
    const int E = in_sizes[9];                   // 800000
    float* out = (float*)d_out;                  // [N,32]

    // ---- workspace carve-up (all f32) ----
    char* ws = (char*)d_ws;
    size_t off = 0;
    auto alloc = [&](size_t elems) -> float* {
        float* p = (float*)(ws + off);
        off += ((elems * sizeof(float)) + 255) & ~(size_t)255;
        return p;
    };
    float* feat1  = alloc((size_t)N * 128);
    float* h1     = alloc((size_t)N * 128);      // out1 accumulator, then h1 in place
    float* el1    = alloc((size_t)N * 4);
    float* er1    = alloc((size_t)N * 4);
    float* emax1  = alloc((size_t)N * 4);
    float* denom1 = alloc((size_t)N * 4);
    float* ee1    = alloc((size_t)E * 4);
    float* feat2  = alloc((size_t)N * 32);
    float* out2   = alloc((size_t)N * 32);
    float* el2    = alloc((size_t)N);
    float* er2    = alloc((size_t)N);
    float* emax2  = alloc((size_t)N);
    float* denom2 = alloc((size_t)N);
    float* ee2    = alloc((size_t)E);
    (void)ws_size;

    const int TB = 256;
    auto blocks = [&](size_t n) { return (int)((n + TB - 1) / TB); };

    // ---- init accumulators (workspace is poisoned between runs) ----
    fill_f32<<<blocks((size_t)N * 4), TB, 0, stream>>>(emax1, -INFINITY, N * 4);
    fill_f32<<<blocks((size_t)N * 4), TB, 0, stream>>>(denom1, 0.f, N * 4);
    fill_f32<<<blocks((size_t)N * 128), TB, 0, stream>>>(h1, 0.f, N * 128);
    fill_f32<<<blocks((size_t)N), TB, 0, stream>>>(emax2, -INFINITY, N);
    fill_f32<<<blocks((size_t)N), TB, 0, stream>>>(denom2, 0.f, N);
    fill_f32<<<blocks((size_t)N * 32), TB, 0, stream>>>(out2, 0.f, N * 32);

    const int mtiles = (N + 15) / 16;

    // ================= Layer 1 (H=4, D=32) =================
    gat_wmma_gemm<256, 128><<<mtiles, 256, 0, stream>>>(h, W1, feat1, N);
    node_attn_proj<<<N, 128, 0, stream>>>(feat1, a_l1, a_r1, el1, er1, 4);
    edge_logit_max<<<blocks(E), TB, 0, stream>>>(el1, er1, src, dst, emax1, E, 4);
    edge_exp_sum<<<blocks(E), TB, 0, stream>>>(el1, er1, src, dst, emax1, ee1, denom1, E, 4);
    edge_scatter<<<E, 128, 0, stream>>>(feat1, ee1, denom1, src, dst, h1, 4);
    elu_bias<<<blocks((size_t)N * 128), TB, 0, stream>>>(h1, b1, N * 128, 128);

    // ================= Layer 2 (H=1, D=32) =================
    gat_wmma_gemm<128, 32><<<mtiles, 64, 0, stream>>>(h1, W2, feat2, N);
    node_attn_proj<<<N, 32, 0, stream>>>(feat2, a_l2, a_r2, el2, er2, 1);
    edge_logit_max<<<blocks(E), TB, 0, stream>>>(el2, er2, src, dst, emax2, E, 1);
    edge_exp_sum<<<blocks(E), TB, 0, stream>>>(el2, er2, src, dst, emax2, ee2, denom2, E, 1);
    edge_scatter<<<E, 32, 0, stream>>>(feat2, ee2, denom2, src, dst, out2, 1);
    add_bias_out<<<blocks((size_t)N * 32), TB, 0, stream>>>(out2, b2, out, N * 32, 32);
}
